// NLinear_63617055588778
// MI455X (gfx1250) — matrix operand order repaired and checked by reading, not compiled, not run on hardware
//
#include <hip/hip_runtime.h>

// ---------------------------------------------------------------------------
// NLinear: out[b,n,o] = sum_i x[b,n,i] * W[n,i,o] + bias[n,o]
// BATCH=4096, N=64, D_IN=512, D_OUT=512, fp32 in/out.
//
// Pass 1 (prep): W[n][i][o] fp32 -> ws as K-transposed bf16 hi/lo Wt[n][o][k]
//                (one 64MB conversion, amortized over 64 reuses per W_n).
// Pass 2 (gemm): 64 batched GEMMs via v_wmma_f32_16x16x32_bf16 with bf16x3
//                error compensation (hi*hi + hi*lo + lo*hi, fp32 accum).
//                B tiles streamed global->LDS by the Tensor Data Mover
//                (tensor_load_to_lds, double buffered, TDM padding gives an
//                80B LDS pitch so fragments are 2x ds_load_b128).
// ---------------------------------------------------------------------------

typedef __attribute__((ext_vector_type(16))) __bf16    v16bf;
typedef __attribute__((ext_vector_type(8)))  float     v8f;
typedef __attribute__((ext_vector_type(4)))  unsigned  u32x4;
typedef __attribute__((ext_vector_type(8)))  int       i32x8;
typedef __attribute__((ext_vector_type(4)))  int       i32x4;

#define BATCH 4096
#define NFEAT 64
#define DIN   512
#define DOUT  512

#define BM 64      // batch rows per block
#define BN 128     // out features per block
#define BK 32      // K step (matches wmma 16x16x32)
#define PADE 40    // LDS row pitch in bf16 elements: 64B data + 16B pad = 80B

// Split fp32 into bf16 hi (RNE, native v_cvt) + bf16 lo (residual, RNE).
__device__ __forceinline__ void split_bf16(float x, unsigned short& hi, unsigned short& lo) {
    __bf16 h = (__bf16)x;            // native f32->bf16 RNE on gfx1250
    float hf = (float)h;
    __bf16 l = (__bf16)(x - hf);
    hi = __builtin_bit_cast(unsigned short, h);
    lo = __builtin_bit_cast(unsigned short, l);
}

union FragBF16 {
    v16bf v;
    u32x4 q[2];
};

// Issue one 2D TDM load: 32(k, contiguous) x 128(o) bf16 tile from Wt[n][o][k]
// into LDS at lds_off.  TDM pad: +4 dwords after every 16 dwords -> 80B pitch.
__device__ __forceinline__ void tdm_load_b_tile(unsigned lds_off, const unsigned short* gptr,
                                                const i32x8& g1, const i32x4& z4) {
    unsigned long long ga = (unsigned long long)gptr;
    u32x4 g0;
    g0[0] = 1u;                                   // count=1, user descriptor
    g0[1] = lds_off;                              // lds_addr (bytes)
    g0[2] = (unsigned)(ga & 0xffffffffu);         // global_addr[31:0]
    g0[3] = (unsigned)(ga >> 32) | (2u << 30);    // global_addr[56:32] | type=2
#if defined(__clang_major__) && (__clang_major__ >= 23)
    i32x8 z8 = {0, 0, 0, 0, 0, 0, 0, 0};
    __builtin_amdgcn_tensor_load_to_lds(g0, g1, z4, z4, z8, 0);
#else
    __builtin_amdgcn_tensor_load_to_lds(g0, g1, z4, z4, 0);
#endif
}

__device__ __forceinline__ i32x8 make_tdm_group1() {
    i32x8 g1;
    g1[0] = (1 << 16)        // data_size = 2 bytes
          | (1 << 20)        // pad_enable
          | (3 << 22)        // pad_interval: 16 dwords (64B of data)
          | (3 << 25);       // pad_amount:   4 dwords (16B)  -> pitch 80B
    g1[1] = (DIN & 0xffff) << 16;                             // tensor_dim0 lo
    g1[2] = ((DIN >> 16) & 0xffff) | ((DOUT & 0xffff) << 16); // dim0 hi | tensor_dim1 lo
    g1[3] = ((DOUT >> 16) & 0xffff) | (BK << 16);             // dim1 hi | tile_dim0 = 32
    g1[4] = BN;                                               // tile_dim1 = 128, tile_dim2 = 0
    g1[5] = DIN;                                              // tensor_dim0_stride = 512
    g1[6] = 0;
    g1[7] = 0;
    return g1;
}

// ---------------------------------------------------------------------------
// Prep: W[n][i][o] fp32 -> Wt_hi[n][o][i], Wt_lo[n][o][i] bf16 (K contiguous)
// ---------------------------------------------------------------------------
__global__ __launch_bounds__(256)
void prep_w_kernel(const float* __restrict__ w,
                   unsigned short* __restrict__ whi,
                   unsigned short* __restrict__ wlo)
{
    __shared__ unsigned short Thi[64][68];
    __shared__ unsigned short Tlo[64][68];

    const int n  = blockIdx.z;
    const int i0 = blockIdx.y * 64;
    const int o0 = blockIdx.x * 64;
    const int tid = threadIdx.x;

    // phase 1: coalesced read of 64(i) x 64(o) fp32 tile, split, stage in LDS [i][o]
    {
        const int o4 = (tid & 15) * 4;
        const int ir = tid >> 4;          // 0..15
        #pragma unroll
        for (int r = 0; r < 4; ++r) {
            const int i = ir + r * 16;
            const float4 q = *(const float4*)(w + ((size_t)n * DIN + i0 + i) * DOUT + o0 + o4);
            unsigned short hh[4], ll[4];
            split_bf16(q.x, hh[0], ll[0]); split_bf16(q.y, hh[1], ll[1]);
            split_bf16(q.z, hh[2], ll[2]); split_bf16(q.w, hh[3], ll[3]);
            uint2 hp, lp;
            hp.x = (unsigned)hh[0] | ((unsigned)hh[1] << 16);
            hp.y = (unsigned)hh[2] | ((unsigned)hh[3] << 16);
            lp.x = (unsigned)ll[0] | ((unsigned)ll[1] << 16);
            lp.y = (unsigned)ll[2] | ((unsigned)ll[3] << 16);
            *(uint2*)&Thi[i][o4] = hp;
            *(uint2*)&Tlo[i][o4] = lp;
        }
    }
    __syncthreads();
    // phase 2: write transposed, i contiguous
    {
        const int i4 = (tid & 15) * 4;
        const int orow = tid >> 4;
        #pragma unroll
        for (int r = 0; r < 4; ++r) {
            const int o = orow + r * 16;
            uint2 hp, lp;
            hp.x = (unsigned)Thi[i4 + 0][o] | ((unsigned)Thi[i4 + 1][o] << 16);
            hp.y = (unsigned)Thi[i4 + 2][o] | ((unsigned)Thi[i4 + 3][o] << 16);
            lp.x = (unsigned)Tlo[i4 + 0][o] | ((unsigned)Tlo[i4 + 1][o] << 16);
            lp.y = (unsigned)Tlo[i4 + 2][o] | ((unsigned)Tlo[i4 + 3][o] << 16);
            const size_t base = ((size_t)n * DOUT + o0 + o) * DIN + i0 + i4;
            *(uint2*)(whi + base) = hp;
            *(uint2*)(wlo + base) = lp;
        }
    }
}

// ---------------------------------------------------------------------------
// Main GEMM
// ---------------------------------------------------------------------------
__global__ __launch_bounds__(256)
void nlinear_bf16x3_wmma(const float* __restrict__ x,
                         const unsigned short* __restrict__ whi,
                         const unsigned short* __restrict__ wlo,
                         const float* __restrict__ bias,
                         float* __restrict__ out)
{
    __shared__ __attribute__((aligned(16))) unsigned short Ahi[BM * PADE];
    __shared__ __attribute__((aligned(16))) unsigned short Alo[BM * PADE];
    __shared__ __attribute__((aligned(16))) unsigned short Bhi[2][BN * PADE];
    __shared__ __attribute__((aligned(16))) unsigned short Blo[2][BN * PADE];

    const int n_idx = blockIdx.z;
    const int b0    = blockIdx.y * BM;
    const int o0    = blockIdx.x * BN;

    const int tid   = threadIdx.x;
    const int lane  = tid & 31;
    const int wave  = tid >> 5;
    const int waveM = wave & 1;
    const int waveN = wave >> 1;

    const int h  = lane >> 4;    // lane half
    const int lr = lane & 15;    // fragment row/col

    const float*          xg  = x   + ((size_t)b0 * NFEAT + n_idx) * DIN;
    const unsigned short* wgh = whi + ((size_t)n_idx * DOUT + o0) * DIN;
    const unsigned short* wgl = wlo + ((size_t)n_idx * DOUT + o0) * DIN;

    v8f acc[2][2];
    acc[0][0] = {}; acc[0][1] = {}; acc[1][0] = {}; acc[1][1] = {};

    // loop-invariant TDM descriptor pieces (stay resident in SGPRs)
    const i32x8 g1 = make_tdm_group1();
    const i32x4 z4 = {0, 0, 0, 0};

    // prologue: TDM prefetch of B tiles for k0 = 0 into buffer 0
    if (wave == 0) {
        tdm_load_b_tile((unsigned)(unsigned long long)(const void*)&Bhi[0][0], wgh, g1, z4);
        tdm_load_b_tile((unsigned)(unsigned long long)(const void*)&Blo[0][0], wgl, g1, z4);
    }

    #pragma unroll 1
    for (int it = 0; it < DIN / BK; ++it) {
        const int k0  = it * BK;
        const int cur = it & 1;

        // ---- stage A tile: 64x32 fp32 -> bf16 hi/lo in LDS [m][k] (pitch 40)
        {
            const int k4   = (tid & 7) * 4;   // 0..28
            const int mrow = tid >> 3;        // 0..31
            #pragma unroll
            for (int half = 0; half < 2; ++half) {
                const int m = mrow + half * 32;
                const float4 q = *(const float4*)(xg + (size_t)m * (NFEAT * DIN) + k0 + k4);
                unsigned short hh[4], ll[4];
                split_bf16(q.x, hh[0], ll[0]); split_bf16(q.y, hh[1], ll[1]);
                split_bf16(q.z, hh[2], ll[2]); split_bf16(q.w, hh[3], ll[3]);
                uint2 hp, lp;
                hp.x = (unsigned)hh[0] | ((unsigned)hh[1] << 16);
                hp.y = (unsigned)hh[2] | ((unsigned)hh[3] << 16);
                lp.x = (unsigned)ll[0] | ((unsigned)ll[1] << 16);
                lp.y = (unsigned)ll[2] | ((unsigned)ll[3] << 16);
                *(uint2*)&Ahi[m * PADE + k4] = hp;
                *(uint2*)&Alo[m * PADE + k4] = lp;
            }
            if (k0 + BK < DIN)
                __builtin_prefetch(xg + (size_t)mrow * (NFEAT * DIN) + k0 + BK, 0, 1);
        }

        // ---- TDM: queue next B tiles, then ensure current tiles landed
        if (wave == 0) {
            if (it + 1 < DIN / BK) {
                const int knext = k0 + BK;
                tdm_load_b_tile((unsigned)(unsigned long long)(const void*)&Bhi[1 - cur][0],
                                wgh + knext, g1, z4);
                tdm_load_b_tile((unsigned)(unsigned long long)(const void*)&Blo[1 - cur][0],
                                wgl + knext, g1, z4);
                __builtin_amdgcn_s_wait_tensorcnt(2);   // current pair complete
            } else {
                __builtin_amdgcn_s_wait_tensorcnt(0);
            }
        }
        __syncthreads();   // A staged + B[cur] visible to all waves

        // ---- gather fragments (ISA 7.12.2 layouts; 16B-aligned -> ds_load_b128)
        FragBF16 fahi[2], falo[2], fbhi[2], fblo[2];
        #pragma unroll
        for (int mi = 0; mi < 2; ++mi) {
            const int m = waveM * 32 + mi * 16 + lr;
            const unsigned short* ph = &Ahi[m * PADE + h * 8];
            const unsigned short* pl = &Alo[m * PADE + h * 8];
            fahi[mi].q[0] = *(const u32x4*)(ph);
            fahi[mi].q[1] = *(const u32x4*)(ph + 16);
            falo[mi].q[0] = *(const u32x4*)(pl);
            falo[mi].q[1] = *(const u32x4*)(pl + 16);
        }
        #pragma unroll
        for (int ni = 0; ni < 2; ++ni) {
            const int o = waveN * 32 + ni * 16 + lr;
            const unsigned short* ph = &Bhi[cur][o * PADE + h * 8];
            const unsigned short* pl = &Blo[cur][o * PADE + h * 8];
            fbhi[ni].q[0] = *(const u32x4*)(ph);
            fbhi[ni].q[1] = *(const u32x4*)(ph + 16);
            fblo[ni].q[0] = *(const u32x4*)(pl);
            fblo[ni].q[1] = *(const u32x4*)(pl + 16);
        }

        // ---- 12 WMMAs: hi*hi + hi*lo + lo*hi, fp32 accumulate
        #pragma unroll
        for (int mi = 0; mi < 2; ++mi) {
            #pragma unroll
            for (int ni = 0; ni < 2; ++ni) {
                acc[mi][ni] = __builtin_amdgcn_wmma_f32_16x16x32_bf16(
                    false, fahi[mi].v, false, fbhi[ni].v, (short)0, acc[mi][ni], false, false);
                acc[mi][ni] = __builtin_amdgcn_wmma_f32_16x16x32_bf16(
                    false, fahi[mi].v, false, fblo[ni].v, (short)0, acc[mi][ni], false, false);
                acc[mi][ni] = __builtin_amdgcn_wmma_f32_16x16x32_bf16(
                    false, falo[mi].v, false, fbhi[ni].v, (short)0, acc[mi][ni], false, false);
            }
        }

        __syncthreads();   // protect A (and B[cur]) before next iteration's writes
    }

    // ---- epilogue: bias add + store (C/D layout: M = r + h*8, N = lane&15)
    #pragma unroll
    for (int ni = 0; ni < 2; ++ni) {
        const int oc = o0 + waveN * 32 + ni * 16 + lr;
        const float bv = bias[(size_t)n_idx * DOUT + oc];
        #pragma unroll
        for (int mi = 0; mi < 2; ++mi) {
            #pragma unroll
            for (int r = 0; r < 8; ++r) {
                const int mrow = b0 + waveM * 32 + mi * 16 + h * 8 + r;
                out[((size_t)mrow * NFEAT + n_idx) * DOUT + oc] = acc[mi][ni][r] + bv;
            }
        }
    }
}

extern "C" void kernel_launch(void* const* d_in, const int* in_sizes, int n_in,
                              void* d_out, int out_size, void* d_ws, size_t ws_size,
                              hipStream_t stream) {
    (void)in_sizes; (void)n_in; (void)out_size; (void)ws_size;
    const float* x    = (const float*)d_in[0];
    const float* wgt  = (const float*)d_in[1];
    const float* bias = (const float*)d_in[2];
    float* out = (float*)d_out;

    unsigned short* whi = (unsigned short*)d_ws;                       // 32 MB
    unsigned short* wlo = whi + (size_t)NFEAT * DIN * DOUT;            // 32 MB

    // Pass 1: convert + transpose W (needs 64 MB of workspace)
    prep_w_kernel<<<dim3(DOUT / 64, DIN / 64, NFEAT), 256, 0, stream>>>(wgt, whi, wlo);

    // Pass 2: WMMA GEMM with TDM-fed B tiles
    nlinear_bf16x3_wmma<<<dim3(DOUT / BN, BATCH / BM, NFEAT), 256, 0, stream>>>(
        x, whi, wlo, bias, out);
}
